// BlockSparseMoE_70660801953900
// MI455X (gfx1250) — compile-verified, matrix operand order checked by reading, not run on hardware
//
#include <hip/hip_runtime.h>

#define T_DIM 2048
#define H_DIM 2048
#define E_DIM 8
#define F_DIM 4096
#define KC 64
#define MT 256          // token tile (M)
#define NT 128          // output tile (N)
#define NTHREADS 512    // 16 waves

#define USE_ASYNC_LDS 1

typedef __attribute__((ext_vector_type(16))) __bf16 v16bf;
typedef __attribute__((ext_vector_type(8)))  __bf16 v8bf;
typedef __attribute__((ext_vector_type(8)))  float  v8f;

struct U128 { unsigned x, y, z, w; };

// ---------- workspace layout ----------
constexpr size_t OFF_TOK   = 0;                                          // E*T int
constexpr size_t OFF_W     = OFF_TOK  + (size_t)E_DIM * T_DIM * 4;       // E*T float
constexpr size_t OFF_CNT   = OFF_W    + (size_t)E_DIM * T_DIM * 4;       // E int (padded)
constexpr size_t OFF_BASE  = OFF_CNT  + 256;                             // E int (padded)
constexpr size_t OFF_XBF   = OFF_BASE + 256;                             // T*H bf16
constexpr size_t OFF_INTER = OFF_XBF  + (size_t)T_DIM * H_DIM * 2;       // (2T)*F bf16

// ---------- helpers ----------
__device__ inline unsigned short bfbits(float f) {
    union { float f; unsigned u; } a; a.f = f;
    unsigned r = a.u + 0x7FFFu + ((a.u >> 16) & 1u);
    return (unsigned short)(r >> 16);
}
__device__ inline unsigned pk2(float lo, float hi) {
    return (unsigned)bfbits(lo) | ((unsigned)bfbits(hi) << 16);
}
__device__ inline v16bf ldfrag(const unsigned short* p) {
    v8bf lo = *(const v8bf*)p;
    v8bf hi = *(const v8bf*)(p + 16);
    return __builtin_shufflevector(lo, hi, 0,1,2,3,4,5,6,7,8,9,10,11,12,13,14,15);
}
__device__ inline void async_cp16(const unsigned short* g, unsigned short* l) {
#if USE_ASYNC_LDS
    unsigned lo = (unsigned)(unsigned long long)(void*)l;   // LDS byte address
    asm volatile("global_load_async_to_lds_b128 %0, %1, off" :: "v"(lo), "v"(g) : "memory");
#else
    *(U128*)l = *(const U128*)g;
#endif
}
__device__ inline void async_wait0() {
#if USE_ASYNC_LDS
#if defined(__has_builtin)
#if __has_builtin(__builtin_amdgcn_s_wait_asynccnt)
    __builtin_amdgcn_s_wait_asynccnt(0);
#else
    asm volatile("s_wait_asynccnt 0" ::: "memory");
#endif
#else
    asm volatile("s_wait_asynccnt 0" ::: "memory");
#endif
#endif
}

// ---------- kernel 0: zero output + counters ----------
__global__ void zero_kernel(float* __restrict__ out, int* __restrict__ counts) {
    size_t i = (size_t)blockIdx.x * blockDim.x + threadIdx.x;
    if (i < (size_t)T_DIM * H_DIM) out[i] = 0.0f;
    if (i < E_DIM) counts[i] = 0;
}

// ---------- kernel 1: x fp32 -> bf16 ----------
__global__ void cvt_x_kernel(const float* __restrict__ x, unsigned* __restrict__ xbf) {
    size_t i = (size_t)blockIdx.x * blockDim.x + threadIdx.x;
    if (i < (size_t)T_DIM * H_DIM / 2)
        xbf[i] = pk2(x[2 * i], x[2 * i + 1]);
}

// ---------- kernel 2: gating + top-2 routing (one wave32 per token) ----------
__global__ void gate_topk_kernel(const float* __restrict__ x,
                                 const float* __restrict__ gw,
                                 int* __restrict__ tok_list,
                                 float* __restrict__ w_list,
                                 int* __restrict__ counts) {
    int wid = threadIdx.x >> 5, lane = threadIdx.x & 31;
    int t = blockIdx.x * 8 + wid;
    if (t >= T_DIM) return;
    float acc[E_DIM];
#pragma unroll
    for (int e = 0; e < E_DIM; ++e) acc[e] = 0.0f;
    for (int h = lane; h < H_DIM; h += 32) {
        float xv = x[(size_t)t * H_DIM + h];
#pragma unroll
        for (int e = 0; e < E_DIM; ++e) acc[e] += xv * gw[e * H_DIM + h];
    }
#pragma unroll
    for (int e = 0; e < E_DIM; ++e)
#pragma unroll
        for (int off = 16; off > 0; off >>= 1) acc[e] += __shfl_down(acc[e], off, 32);
    if (lane == 0) {
        float mx = acc[0];
#pragma unroll
        for (int e = 1; e < E_DIM; ++e) mx = fmaxf(mx, acc[e]);
        float p[E_DIM], s = 0.0f;
#pragma unroll
        for (int e = 0; e < E_DIM; ++e) { p[e] = __expf(acc[e] - mx); s += p[e]; }
        float inv = 1.0f / s;
#pragma unroll
        for (int e = 0; e < E_DIM; ++e) p[e] *= inv;
        int i0 = 0;
#pragma unroll
        for (int e = 1; e < E_DIM; ++e) if (p[e] > p[i0]) i0 = e;   // first index wins ties
        int i1 = (i0 == 0) ? 1 : 0;
#pragma unroll
        for (int e = 0; e < E_DIM; ++e) if (e != i0 && p[e] > p[i1] && !(e > i1 && p[e] == p[i1])) i1 = e;
        float s2 = 1.0f / (p[i0] + p[i1]);
        int s0 = atomicAdd(&counts[i0], 1);
        tok_list[i0 * T_DIM + s0] = t;  w_list[i0 * T_DIM + s0] = p[i0] * s2;
        int s1 = atomicAdd(&counts[i1], 1);
        tok_list[i1 * T_DIM + s1] = t;  w_list[i1 * T_DIM + s1] = p[i1] * s2;
    }
}

// ---------- kernel 3: per-expert row-base prefix sum ----------
__global__ void prefix_kernel(const int* __restrict__ counts, int* __restrict__ row_base) {
    if (threadIdx.x == 0 && blockIdx.x == 0) {
        int s = 0;
        for (int e = 0; e < E_DIM; ++e) { row_base[e] = s; s += counts[e]; }
    }
}

// ---------- kernel 4: up-projection GEMMs + SwiGLU ----------
__global__ __launch_bounds__(NTHREADS) void ffn_up_kernel(
        const unsigned short* __restrict__ xbf,
        const float* __restrict__ w1, const float* __restrict__ w3,
        const int* __restrict__ tok_list, const int* __restrict__ counts,
        const int* __restrict__ row_base, unsigned short* __restrict__ inter) {
    int e = blockIdx.z, mtile = blockIdx.y, ftile = blockIdx.x;
    int ne = counts[e];
    if (mtile * MT >= ne) return;

    __shared__ unsigned short sA[MT * KC];     // 32 KB
    __shared__ unsigned short sB1[NT * KC];    // 16 KB
    __shared__ unsigned short sB3[NT * KC];    // 16 KB

    int tid = threadIdx.x, lane = tid & 31, wid = tid >> 5;
    int mbase = (wid & 7) * 32, nbase = (wid >> 3) * 64;

    // per-thread A-staging source rows (fixed across K loop): 4 x 16B chunks
    const unsigned short* arow[4];
    unsigned short* adst[4];
#pragma unroll
    for (int it = 0; it < 4; ++it) {
        int idx = tid + it * NTHREADS;          // 0..2047
        int row = idx >> 3, q = idx & 7;
        int gr = mtile * MT + row;
        int t = (gr < ne) ? tok_list[e * T_DIM + gr] : 0;   // tail rows masked in epilogue
        arow[it] = xbf + (size_t)t * H_DIM + q * 8;
        adst[it] = sA + row * KC + q * 8;
    }

    v8f acc1[2][4], acc3[2][4];
#pragma unroll
    for (int mi = 0; mi < 2; ++mi)
#pragma unroll
        for (int ni = 0; ni < 4; ++ni) { acc1[mi][ni] = (v8f)(0.0f); acc3[mi][ni] = (v8f)(0.0f); }

    const float* w1e = w1 + ((size_t)e * F_DIM + (size_t)ftile * NT) * H_DIM;
    const float* w3e = w3 + ((size_t)e * F_DIM + (size_t)ftile * NT) * H_DIM;

    for (int kk = 0; kk < H_DIM; kk += KC) {
        __syncthreads();
        // stage A (tokens x KC) via async copy to LDS
#pragma unroll
        for (int it = 0; it < 4; ++it) async_cp16(arow[it] + kk, adst[it]);
        // stage B1/B3 (F-rows x KC), fp32 -> bf16 on the fly; prefetch next chunk
#pragma unroll
        for (int it = 0; it < 4; ++it) {
            int idx = tid + it * NTHREADS;      // 0..2047
            int row = idx >> 4, q = idx & 15;
            const float* p1 = w1e + (size_t)row * H_DIM + kk + q * 4;
            const float* p3 = w3e + (size_t)row * H_DIM + kk + q * 4;
            if (kk + KC < H_DIM) {
                __builtin_prefetch(p1 + KC, 0, 0);
                __builtin_prefetch(p3 + KC, 0, 0);
            }
            const float4 a = *(const float4*)p1;
            const float4 b = *(const float4*)p3;
            unsigned* d1 = (unsigned*)(sB1 + row * KC + q * 4);
            unsigned* d3 = (unsigned*)(sB3 + row * KC + q * 4);
            d1[0] = pk2(a.x, a.y); d1[1] = pk2(a.z, a.w);
            d3[0] = pk2(b.x, b.y); d3[1] = pk2(b.z, b.w);
        }
        async_wait0();
        __syncthreads();
#pragma unroll
        for (int ks = 0; ks < KC; ks += 32) {
            int klo = ks + (lane >> 4) * 8;
            v16bf afr[2];
#pragma unroll
            for (int mi = 0; mi < 2; ++mi)
                afr[mi] = ldfrag(sA + (mbase + mi * 16 + (lane & 15)) * KC + klo);
#pragma unroll
            for (int ni = 0; ni < 4; ++ni) {
                int col = nbase + ni * 16 + (lane & 15);
                v16bf b1 = ldfrag(sB1 + col * KC + klo);
                v16bf b3 = ldfrag(sB3 + col * KC + klo);
#pragma unroll
                for (int mi = 0; mi < 2; ++mi) {
                    acc1[mi][ni] = __builtin_amdgcn_wmma_f32_16x16x32_bf16(
                        false, afr[mi], false, b1, (short)0, acc1[mi][ni], false, false);
                    acc3[mi][ni] = __builtin_amdgcn_wmma_f32_16x16x32_bf16(
                        false, afr[mi], false, b3, (short)0, acc3[mi][ni], false, false);
                }
            }
        }
    }
    // epilogue: SwiGLU, store compact bf16 intermediate
    int rb = row_base[e];
#pragma unroll
    for (int mi = 0; mi < 2; ++mi)
#pragma unroll
        for (int j = 0; j < 8; ++j) {
            int rl = mbase + mi * 16 + j + ((lane >> 4) << 3);
            int gr = mtile * MT + rl;
            if (gr < ne) {
#pragma unroll
                for (int ni = 0; ni < 4; ++ni) {
                    float h1 = acc1[mi][ni][j], h3 = acc3[mi][ni][j];
                    float v = (h1 / (1.0f + __expf(-h1))) * h3;
                    int f = ftile * NT + nbase + ni * 16 + (lane & 15);
                    inter[(size_t)(rb + gr) * F_DIM + f] = bfbits(v);
                }
            }
        }
}

// ---------- kernel 5: down-projection GEMM + weighted scatter ----------
__global__ __launch_bounds__(NTHREADS) void ffn_down_kernel(
        const unsigned short* __restrict__ inter, const float* __restrict__ w2,
        const int* __restrict__ tok_list, const float* __restrict__ w_list,
        const int* __restrict__ counts, const int* __restrict__ row_base,
        float* __restrict__ out) {
    int e = blockIdx.z, mtile = blockIdx.y, htile = blockIdx.x;
    int ne = counts[e];
    if (mtile * MT >= ne) return;

    __shared__ unsigned short sA[MT * KC];     // 32 KB
    __shared__ unsigned short sB[NT * KC];     // 16 KB  [h][f]
    __shared__ int   sTok[MT];
    __shared__ float sW[MT];

    int tid = threadIdx.x, lane = tid & 31, wid = tid >> 5;
    int mbase = (wid & 7) * 32, nbase = (wid >> 3) * 64;
    int rb = row_base[e];

    if (tid < MT) {
        int gr = mtile * MT + tid;
        sTok[tid] = (gr < ne) ? tok_list[e * T_DIM + gr] : -1;
        sW[tid]   = (gr < ne) ? w_list[e * T_DIM + gr] : 0.0f;
    }

    const unsigned short* arow[4];
    unsigned short* adst[4];
#pragma unroll
    for (int it = 0; it < 4; ++it) {
        int idx = tid + it * NTHREADS;
        int row = idx >> 3, q = idx & 7;
        int gr = mtile * MT + row;
        size_t rr = (gr < ne) ? (size_t)(rb + gr) : 0;     // tail rows masked in epilogue
        arow[it] = inter + rr * F_DIM + q * 8;
        adst[it] = sA + row * KC + q * 8;
    }

    v8f acc[2][4];
#pragma unroll
    for (int mi = 0; mi < 2; ++mi)
#pragma unroll
        for (int ni = 0; ni < 4; ++ni) acc[mi][ni] = (v8f)(0.0f);

    for (int kk = 0; kk < F_DIM; kk += KC) {
        __syncthreads();
#pragma unroll
        for (int it = 0; it < 4; ++it) async_cp16(arow[it] + kk, adst[it]);
        // stage B: w2 rows (f, h) contiguous in h; transpose to sB[h][f]
#pragma unroll
        for (int it = 0; it < 4; ++it) {
            int idx = tid + it * NTHREADS;      // 64 f-rows x 32 h-quads
            int f = idx >> 5, hq = idx & 31;
            const float* p = w2 + ((size_t)e * F_DIM + kk + f) * H_DIM
                             + (size_t)htile * NT + hq * 4;
            if (kk + KC < F_DIM) __builtin_prefetch(p + (size_t)KC * H_DIM, 0, 0);
            const float4 a = *(const float4*)p;
            sB[(hq * 4 + 0) * KC + f] = bfbits(a.x);
            sB[(hq * 4 + 1) * KC + f] = bfbits(a.y);
            sB[(hq * 4 + 2) * KC + f] = bfbits(a.z);
            sB[(hq * 4 + 3) * KC + f] = bfbits(a.w);
        }
        async_wait0();
        __syncthreads();
#pragma unroll
        for (int ks = 0; ks < KC; ks += 32) {
            int klo = ks + (lane >> 4) * 8;
            v16bf afr[2];
#pragma unroll
            for (int mi = 0; mi < 2; ++mi)
                afr[mi] = ldfrag(sA + (mbase + mi * 16 + (lane & 15)) * KC + klo);
#pragma unroll
            for (int ni = 0; ni < 4; ++ni) {
                v16bf bfr = ldfrag(sB + (nbase + ni * 16 + (lane & 15)) * KC + klo);
#pragma unroll
                for (int mi = 0; mi < 2; ++mi)
                    acc[mi][ni] = __builtin_amdgcn_wmma_f32_16x16x32_bf16(
                        false, afr[mi], false, bfr, (short)0, acc[mi][ni], false, false);
            }
        }
    }
    // epilogue: apply routing weight, accumulate into output
#pragma unroll
    for (int mi = 0; mi < 2; ++mi)
#pragma unroll
        for (int j = 0; j < 8; ++j) {
            int rl = mbase + mi * 16 + j + ((lane >> 4) << 3);
            int gr = mtile * MT + rl;
            if (gr < ne) {
                int t = sTok[rl];
                float wt = sW[rl];
#pragma unroll
                for (int ni = 0; ni < 4; ++ni) {
                    int h = htile * NT + nbase + ni * 16 + (lane & 15);
                    atomicAdd(&out[(size_t)t * H_DIM + h], wt * acc[mi][ni][j]);
                }
            }
        }
}

extern "C" void kernel_launch(void* const* d_in, const int* in_sizes, int n_in,
                              void* d_out, int out_size, void* d_ws, size_t ws_size,
                              hipStream_t stream) {
    const float* x  = (const float*)d_in[0];
    const float* gw = (const float*)d_in[1];
    const float* w1 = (const float*)d_in[2];
    const float* w2 = (const float*)d_in[3];
    const float* w3 = (const float*)d_in[4];
    float* out = (float*)d_out;
    char* ws = (char*)d_ws;

    int*            tok   = (int*)(ws + OFF_TOK);
    float*          wl    = (float*)(ws + OFF_W);
    int*            cnt   = (int*)(ws + OFF_CNT);
    int*            base  = (int*)(ws + OFF_BASE);
    unsigned short* xbf   = (unsigned short*)(ws + OFF_XBF);
    unsigned short* inter = (unsigned short*)(ws + OFF_INTER);

    zero_kernel<<<(T_DIM * H_DIM + 255) / 256, 256, 0, stream>>>(out, cnt);
    cvt_x_kernel<<<(T_DIM * H_DIM / 2 + 255) / 256, 256, 0, stream>>>(x, (unsigned*)xbf);
    gate_topk_kernel<<<T_DIM / 8, 256, 0, stream>>>(x, gw, tok, wl, cnt);
    prefix_kernel<<<1, 1, 0, stream>>>(cnt, base);

    dim3 g4(F_DIM / NT, T_DIM / MT, E_DIM);
    ffn_up_kernel<<<g4, NTHREADS, 0, stream>>>(xbf, w1, w3, tok, cnt, base, inter);

    dim3 g5(H_DIM / NT, T_DIM / MT, E_DIM);
    ffn_down_kernel<<<g5, NTHREADS, 0, stream>>>(inter, w2, tok, wl, cnt, base, out);
}
